// network_52149492908558
// MI455X (gfx1250) — matrix operand order repaired and checked by reading, not compiled, not run on hardware
//
#include <hip/hip_runtime.h>
#include <hip/hip_bf16.h>
#include <cstdint>
#include <cstddef>

typedef __attribute__((ext_vector_type(16))) _Float16 v16h;
typedef __attribute__((ext_vector_type(8)))  float    v8f;

#define BN_EPS 1e-5f
#define CONV_CW 128   // output x positions per block chunk

__device__ __forceinline__ int imin(int a, int b){ return a < b ? a : b; }
__device__ __forceinline__ int imax(int a, int b){ return a > b ? a : b; }

// ---------------- elementwise f32 -> f16 cast ----------------
__global__ __launch_bounds__(256)
void k_cast_f32_to_f16(const float* __restrict__ in, _Float16* __restrict__ out, size_t n){
  size_t i  = (size_t)blockIdx.x * blockDim.x + threadIdx.x;
  size_t st = (size_t)gridDim.x * blockDim.x;
  for (; i < n; i += st) out[i] = (_Float16)in[i];
}

// ---------------- weight prepack ----------------
// src: [Cout][Cin][27] f32  ->  dst: [Coutp][27*Cinp] f16, zero padded.
// K ordering: k = tap*Cinp + ci  (tap = (kd*3+kh)*3+kw)
__global__ __launch_bounds__(256)
void k_pack_w(const float* __restrict__ src, _Float16* __restrict__ dst,
              int Cin, int Cinp, int Cout, int Coutp){
  const int Kp = 27 * Cinp;
  size_t n  = (size_t)Coutp * Kp;
  size_t i  = (size_t)blockIdx.x * blockDim.x + threadIdx.x;
  size_t st = (size_t)gridDim.x * blockDim.x;
  for (; i < n; i += st){
    int co = (int)(i / Kp);
    int r  = (int)(i % Kp);
    int t  = r / Cinp;
    int ci = r % Cinp;
    float v = 0.0f;
    if (co < Cout && ci < Cin) v = src[((size_t)co * Cin + ci) * 27 + t];
    dst[i] = (_Float16)v;
  }
}

// ---------------- implicit-GEMM 3x3x3 conv via WMMA ----------------
// X:  fp16 activations [Cin][Din][Hin][Win]
// Wp: fp16 prepacked weights [Coutp][27*Cinp]
// Y:  fp32 pre-BN output [Cout][Dout][Hout][Wout]
//
// Block: 256 thr / 8 waves. One block = one output (z,y) row x-chunk of
// CONV_CW outputs; wave w owns the 16-voxel N-tile starting at w*16.
// grid.y = cout tile. Per kd: stage 3 (kh) input rows (all padded channels,
// channel-innermost, zero-filled halo) into LDS, then 3x3x(Cinp/32) WMMAs.
// LDS channel pitch = Cinp+8 halves: keeps 16B alignment for ds_load_b128
// while spreading the per-column bank stride (20/36 banks).
__global__ __launch_bounds__(256)
void k_conv3d_wmma(const _Float16* __restrict__ X, const _Float16* __restrict__ Wp,
                   float* __restrict__ Y,
                   int Cin, int Cinp, int Cout,
                   int Din, int Hin, int Win,
                   int Dout, int Hout, int Wout,
                   int stride, int nchunks)
{
  __shared__ _Float16 lds[30960]; // 61,920 B: max(3*258*40, 3*130*72)

  const int Kp    = 27 * Cinp;
  const int XL    = CONV_CW * stride + 2;  // staged x extent
  const int CinpL = Cinp + 8;              // padded LDS channel pitch

  const int chunk = blockIdx.x % nchunks;
  const int row   = blockIdx.x / nchunks;  // zo*Hout + yo
  const int zo    = row / Hout;
  const int yo    = row - zo * Hout;
  const int cotile = blockIdx.y;

  const int xbase = chunk * CONV_CW;
  const int xmin  = xbase * stride - 1;    // global x of LDS x-index 0

  const int lane = threadIdx.x & 31;
  const int wave = threadIdx.x >> 5;
  const int half = lane >> 4;
  const int lo16 = lane & 15;
  const int t16  = wave * 16;              // tile base inside chunk

  const size_t planeIn = (size_t)Hin * Win;
  const size_t volIn   = (size_t)Din * planeIn;

  const bool active = (xbase + t16) < Wout;   // wave-uniform
  const _Float16* wbase = Wp + ((size_t)cotile * 16 + lo16) * Kp;

  v8f acc = {};

  for (int kd = 0; kd < 3; ++kd){
    const int zi  = zo * stride + kd - 1;
    const bool zok = (unsigned)zi < (unsigned)Din;

    __syncthreads();  // protect LDS from previous iteration's readers
    { // cooperative staging: idx = (kh*Cinp + ci)*XL + x  (x fastest -> coalesced)
      const int total = 3 * Cinp * XL;
      for (int idx = threadIdx.x; idx < total; idx += 256){
        const int x  = idx % XL;
        const int r  = idx / XL;
        const int ci = r % Cinp;
        const int kh = r / Cinp;
        const int yi = yo * stride + kh - 1;
        const int gx = xmin + x;
        _Float16 v = (_Float16)0.0f;
        if (zok && (unsigned)yi < (unsigned)Hin &&
            (unsigned)gx < (unsigned)Win && ci < Cin)
          v = X[(size_t)ci * volIn + (size_t)zi * planeIn + (size_t)yi * Win + gx];
        lds[(size_t)(kh * XL + x) * CinpL + ci] = v;
      }
    }
    __syncthreads();

    if (active){
      #pragma unroll
      for (int kh = 0; kh < 3; ++kh){
        #pragma unroll
        for (int kw = 0; kw < 3; ++kw){
          const int xl = (t16 + lo16) * stride + kw;   // LDS x for this lane's col
          for (int cc = 0; cc < Cinp; cc += 32){
            const int kbase = ((kd * 3 + kh) * 3 + kw) * Cinp + cc;

            // A fragment (16x32 f16): row M = lo16;
            // elem j -> K = half*16+j (matches B fragment order)
            v16h a;
            const _Float16* wrow = wbase + kbase;
            #pragma unroll
            for (int j = 0; j < 16; ++j){
              const int kk = half * 16 + j;
              a[j] = wrow[kk];
            }

            // B fragment (32x16 f16): col N = lo16, elem j -> K = half*16+j;
            // contiguous 32 bytes in LDS -> 2x ds_load_b128.
            v16h b;
            const _Float16* lrow = &lds[(size_t)(kh * XL + xl) * CinpL + cc + half * 16];
            #pragma unroll
            for (int j = 0; j < 16; ++j) b[j] = lrow[j];

            acc = __builtin_amdgcn_wmma_f32_16x16x32_f16(
                false, a, false, b, (short)0, acc, false, false);
          }
        }
      }
    }
  }

  if (active){
    // C/D layout: VGPR r -> M = half*8 + r, column N = lo16  [ISA 7.12.2]
    const size_t planeOut = (size_t)Hout * Wout;
    const size_t volOut   = (size_t)Dout * planeOut;
    const int xout = xbase + t16 + lo16;
    if (xout < Wout){
      float* Yzy = Y + (size_t)zo * planeOut + (size_t)yo * Wout + xout;
      #pragma unroll
      for (int r = 0; r < 8; ++r){
        const int co = cotile * 16 + half * 8 + r;
        if (co < Cout) Yzy[(size_t)co * volOut] = acc[r];
      }
    }
  }
}

// ---------------- BatchNorm statistics ----------------
__global__ void k_zero(float* p, int n){
  int i = blockIdx.x * blockDim.x + threadIdx.x;
  if (i < n) p[i] = 0.0f;
}

__global__ __launch_bounds__(256)
void k_bn_partial(const float* __restrict__ Y, float* __restrict__ sums, size_t N){
  const int c = blockIdx.x;
  __shared__ float s1[256], s2[256];
  const float* Yc = Y + (size_t)c * N;
  float a = 0.0f, b = 0.0f;
  for (size_t i = (size_t)blockIdx.y * blockDim.x + threadIdx.x; i < N;
       i += (size_t)gridDim.y * blockDim.x){
    float v = Yc[i];
    a += v;
    b += v * v;
  }
  s1[threadIdx.x] = a; s2[threadIdx.x] = b;
  __syncthreads();
  for (int o = 128; o > 0; o >>= 1){
    if ((int)threadIdx.x < o){
      s1[threadIdx.x] += s1[threadIdx.x + o];
      s2[threadIdx.x] += s2[threadIdx.x + o];
    }
    __syncthreads();
  }
  if (threadIdx.x == 0){
    atomicAdd(&sums[2 * c + 0], s1[0]);
    atomicAdd(&sums[2 * c + 1], s2[0]);
  }
}

__global__ void k_bn_finalize(const float* __restrict__ sums,
                              const float* __restrict__ g, const float* __restrict__ b,
                              float* __restrict__ ss, int C, float invN){
  int c = blockIdx.x * blockDim.x + threadIdx.x;
  if (c >= C) return;
  float mean = sums[2 * c] * invN;
  float var  = sums[2 * c + 1] * invN - mean * mean;
  float sc   = g[c] * rsqrtf(var + BN_EPS);
  ss[2 * c + 0] = sc;
  ss[2 * c + 1] = b[c] - mean * sc;
}

// y = relu(conv*scale+shift) (+ skip), cast to fp16.  grid.y = channel.
__global__ __launch_bounds__(256)
void k_bn_relu(const float* __restrict__ Y, const float* __restrict__ ss,
               const _Float16* __restrict__ skip, _Float16* __restrict__ out,
               size_t N){
  const int c = blockIdx.y;
  const float sc = ss[2 * c], sh = ss[2 * c + 1];
  const size_t base = (size_t)c * N;
  size_t i  = (size_t)blockIdx.x * blockDim.x + threadIdx.x;
  size_t st = (size_t)gridDim.x * blockDim.x;
  for (; i < N; i += st){
    float v = fmaxf(Y[base + i] * sc + sh, 0.0f);
    if (skip) v += (float)skip[base + i];
    out[base + i] = (_Float16)v;
  }
}

// ---------------- trilinear x2 upsample (align_corners=False) ----------------
// grid.y = channel
__global__ __launch_bounds__(256)
void k_up2(const _Float16* __restrict__ in, _Float16* __restrict__ out,
           int D, int H, int W){
  const int Do = 2 * D, Ho = 2 * H, Wo = 2 * W;
  const int c = blockIdx.y;
  const _Float16* pc = in + (size_t)c * D * H * W;
  _Float16* oc = out + (size_t)c * Do * Ho * Wo;
  const size_t total = (size_t)Do * Ho * Wo;
  size_t i  = (size_t)blockIdx.x * blockDim.x + threadIdx.x;
  size_t st = (size_t)gridDim.x * blockDim.x;
  for (; i < total; i += st){
    size_t t = i;
    const int xo = (int)(t % Wo); t /= Wo;
    const int yo = (int)(t % Ho); t /= Ho;
    const int zo = (int)t;
    const float fx = (xo + 0.5f) * 0.5f - 0.5f;
    const float fy = (yo + 0.5f) * 0.5f - 0.5f;
    const float fz = (zo + 0.5f) * 0.5f - 0.5f;
    const int x0 = (int)floorf(fx), y0 = (int)floorf(fy), z0 = (int)floorf(fz);
    const float wx = fx - x0, wy = fy - y0, wz = fz - z0;
    const int x0c = imax(x0, 0), x1c = imin(x0 + 1, W - 1);
    const int y0c = imax(y0, 0), y1c = imin(y0 + 1, H - 1);
    const int z0c = imax(z0, 0), z1c = imin(z0 + 1, D - 1);
    #define AT(z, y, x) ((float)pc[((size_t)(z) * H + (y)) * W + (x)])
    const float v000 = AT(z0c, y0c, x0c), v001 = AT(z0c, y0c, x1c);
    const float v010 = AT(z0c, y1c, x0c), v011 = AT(z0c, y1c, x1c);
    const float v100 = AT(z1c, y0c, x0c), v101 = AT(z1c, y0c, x1c);
    const float v110 = AT(z1c, y1c, x0c), v111 = AT(z1c, y1c, x1c);
    #undef AT
    const float v00 = v000 + (v001 - v000) * wx;
    const float v01 = v010 + (v011 - v010) * wx;
    const float v10 = v100 + (v101 - v100) * wx;
    const float v11 = v110 + (v111 - v110) * wx;
    const float v0  = v00 + (v01 - v00) * wy;
    const float v1  = v10 + (v11 - v10) * wy;
    oc[i] = (_Float16)(v0 + (v1 - v0) * wz);
  }
}

// ---------------- final 8->1 conv + bias (fp32 out) ----------------
__global__ __launch_bounds__(256)
void k_prob(const _Float16* __restrict__ X, const float* __restrict__ w,
            const float* __restrict__ bias, float* __restrict__ out,
            int D, int H, int W){
  size_t total = (size_t)D * H * W;
  size_t i  = (size_t)blockIdx.x * blockDim.x + threadIdx.x;
  size_t st = (size_t)gridDim.x * blockDim.x;
  for (; i < total; i += st){
    size_t t = i;
    const int x = (int)(t % W); t /= W;
    const int y = (int)(t % H); t /= H;
    const int z = (int)t;
    float acc = bias[0];
    for (int ci = 0; ci < 8; ++ci){
      const _Float16* pc = X + (size_t)ci * D * H * W;
      for (int kd = 0; kd < 3; ++kd){
        const int zi = z + kd - 1;
        if ((unsigned)zi >= (unsigned)D) continue;
        for (int kh = 0; kh < 3; ++kh){
          const int yi = y + kh - 1;
          if ((unsigned)yi >= (unsigned)H) continue;
          for (int kw = 0; kw < 3; ++kw){
            const int xi = x + kw - 1;
            if ((unsigned)xi >= (unsigned)W) continue;
            acc += w[ci * 27 + kd * 9 + kh * 3 + kw] *
                   (float)pc[((size_t)zi * H + yi) * W + xi];
          }
        }
      }
    }
    out[i] = acc;
  }
}

// ---------------- host orchestration ----------------
extern "C" void kernel_launch(void* const* d_in, const int* in_sizes, int n_in,
                              void* d_out, int out_size, void* d_ws, size_t ws_size,
                              hipStream_t stream){
  (void)in_sizes; (void)n_in; (void)out_size; (void)ws_size;

  static const int LD[4] = {64, 32, 16, 8};
  static const int LH[4] = {160, 80, 40, 20};
  static const int LW[4] = {192, 96, 48, 24};
  struct Lr { int cin, cout, stride, inL, outL; };
  static const Lr L[14] = {
    {8, 8, 1, 0, 0},  {8, 8, 1, 0, 0},                       // input0, input1
    {8, 16, 2, 0, 1}, {16, 16, 1, 1, 1}, {16, 16, 1, 1, 1},  // conv1a..c
    {16, 32, 2, 1, 2}, {32, 32, 1, 2, 2}, {32, 32, 1, 2, 2}, // conv2a..c
    {32, 64, 2, 2, 3}, {64, 64, 1, 3, 3}, {64, 64, 1, 3, 3}, // conv3a..c
    {64, 32, 1, 2, 2},                                       // conv3d (on up2(c3))
    {32, 16, 1, 1, 1},                                       // conv2d (on up2(d3))
    {16, 8, 1, 0, 0}};                                       // conv1d (on up2(d2))

  const size_t V0 = (size_t)64 * 160 * 192;

  char* ws = (char*)d_ws;
  size_t off = 0;
  auto arena = [&](size_t bytes) -> char* {
    char* p = ws + off;
    off = (off + bytes + 255) & ~(size_t)255;
    return p;
  };
  _Float16* A    = (_Float16*)arena(16 * V0 * sizeof(_Float16)); // ping
  _Float16* B    = (_Float16*)arena(16 * V0 * sizeof(_Float16)); // pong
  float*    S    = (float*)arena(8 * V0 * sizeof(float));        // pre-BN staging
  _Float16* i0b  = (_Float16*)arena(8 * V0 * sizeof(_Float16));  // skip i0
  _Float16* c1b  = (_Float16*)arena((size_t)16 * 32 * 80 * 96 * sizeof(_Float16));
  _Float16* c2b  = (_Float16*)arena((size_t)32 * 16 * 40 * 48 * sizeof(_Float16));
  float*    sums = (float*)arena(1024);
  float*    ssb  = (float*)arena(1024);

  // prepack all conv weights to fp16 [Coutp][27*Cinp]
  size_t woff[14]; size_t wtot = 0;
  for (int i = 0; i < 14; ++i){
    const int cinp    = (L[i].cin <= 32) ? 32 : 64;
    const int cotiles = (L[i].cout + 15) / 16;
    woff[i] = wtot;
    wtot += (size_t)cotiles * 16 * 27 * cinp;
  }
  _Float16* Wp = (_Float16*)arena(wtot * sizeof(_Float16));
  for (int i = 0; i < 14; ++i){
    const int cinp    = (L[i].cin <= 32) ? 32 : 64;
    const int cotiles = (L[i].cout + 15) / 16;
    const size_t n    = (size_t)cotiles * 16 * 27 * cinp;
    k_pack_w<<<dim3((unsigned)((n + 255) / 256)), 256, 0, stream>>>(
        (const float*)d_in[1 + 3 * i], Wp + woff[i], L[i].cin, cinp, L[i].cout, cotiles * 16);
  }

  // cast network input x -> fp16 in A
  k_cast_f32_to_f16<<<2048, 256, 0, stream>>>((const float*)d_in[0], A, 8 * V0);

  auto cbr = [&](const _Float16* in, _Float16* outp, int li, const _Float16* skip){
    const Lr& l = L[li];
    const int Din = LD[l.inL],  Hin = LH[l.inL],  Win = LW[l.inL];
    const int Dout = LD[l.outL], Hout = LH[l.outL], Wout = LW[l.outL];
    const int cinp    = (l.cin <= 32) ? 32 : 64;
    const int cotiles = (l.cout + 15) / 16;
    const int nchunks = (Wout + CONV_CW - 1) / CONV_CW;
    const size_t N = (size_t)Dout * Hout * Wout;

    k_conv3d_wmma<<<dim3((unsigned)(Dout * Hout * nchunks), (unsigned)cotiles),
                    256, 0, stream>>>(
        in, Wp + woff[li], S, l.cin, cinp, l.cout,
        Din, Hin, Win, Dout, Hout, Wout, l.stride, nchunks);

    k_zero<<<1, 128, 0, stream>>>(sums, 2 * l.cout);
    unsigned nb = (unsigned)((N + 262143) / 262144);
    if (nb < 1) nb = 1; if (nb > 256) nb = 256;
    k_bn_partial<<<dim3((unsigned)l.cout, nb), 256, 0, stream>>>(S, sums, N);
    k_bn_finalize<<<1, 64, 0, stream>>>(
        sums, (const float*)d_in[2 + 3 * li], (const float*)d_in[3 + 3 * li],
        ssb, l.cout, 1.0f / (float)N);

    unsigned gb = (unsigned)((N + 255) / 256);
    if (gb > 8192u) gb = 8192u;
    k_bn_relu<<<dim3(gb, (unsigned)l.cout), 256, 0, stream>>>(S, ssb, skip, outp, N);
  };

  // ---- encoder ----
  cbr(A, B, 0, nullptr);       // input0  -> B   (8 @ L0)
  cbr(B, i0b, 1, nullptr);     // input1  -> i0  (8 @ L0, skip)
  cbr(i0b, A, 2, nullptr);     // conv1a  -> A   (16 @ L1)
  cbr(A, B, 3, nullptr);       // conv1b  -> B
  cbr(B, c1b, 4, nullptr);     // conv1c  -> c1  (16 @ L1, skip)
  cbr(c1b, A, 5, nullptr);     // conv2a  -> A   (32 @ L2)
  cbr(A, B, 6, nullptr);       // conv2b  -> B
  cbr(B, c2b, 7, nullptr);     // conv2c  -> c2  (32 @ L2, skip)
  cbr(c2b, A, 8, nullptr);     // conv3a  -> A   (64 @ L3)
  cbr(A, B, 9, nullptr);       // conv3b  -> B
  cbr(B, A, 10, nullptr);      // conv3c  -> A = c3 (64 @ L3)

  // ---- decoder ----
  { // up2(c3): 64ch L3 -> L2, into B
    const size_t n = (size_t)LD[2] * LH[2] * LW[2];
    unsigned gb = (unsigned)((n + 255) / 256);
    k_up2<<<dim3(gb, 64), 256, 0, stream>>>(A, B, LD[3], LH[3], LW[3]);
  }
  cbr(B, A, 11, c2b);          // d3 = c2 + cbr(up2(c3))  -> A (32 @ L2)
  { // up2(d3): 32ch L2 -> L1, into B
    const size_t n = (size_t)LD[1] * LH[1] * LW[1];
    unsigned gb = (unsigned)((n + 255) / 256);
    k_up2<<<dim3(gb, 32), 256, 0, stream>>>(A, B, LD[2], LH[2], LW[2]);
  }
  cbr(B, A, 12, c1b);          // d2 = c1 + cbr(up2(d3))  -> A (16 @ L1)
  { // up2(d2): 16ch L1 -> L0, into B
    const size_t n = (size_t)LD[0] * LH[0] * LW[0];
    unsigned gb = (unsigned)((n + 255) / 256);
    if (gb > 16384u) gb = 16384u;
    k_up2<<<dim3(gb, 16), 256, 0, stream>>>(A, B, LD[1], LH[1], LW[1]);
  }
  cbr(B, A, 13, i0b);          // d1 = i0 + cbr(up2(d2))  -> A (8 @ L0)

  { // prob0: 8 -> 1 conv + bias, fp32 output
    unsigned gb = (unsigned)((V0 + 255) / 256);
    if (gb > 65535u) gb = 65535u;
    k_prob<<<gb, 256, 0, stream>>>(A, (const float*)d_in[43], (const float*)d_in[44],
                                   (float*)d_out, LD[0], LH[0], LW[0]);
  }
}
